// RegionProposalNetwork_55748675502388
// MI455X (gfx1250) — matrix operand order repaired and checked by reading, not compiled, not run on hardware
//
#include <hip/hip_runtime.h>
#include <hip/hip_bf16.h>
#include <math.h>

typedef __bf16 bf16;
typedef __attribute__((ext_vector_type(16))) __bf16 v16bf;
typedef __attribute__((ext_vector_type(8)))  float  v8f;

#define BATCH   8
#define CIN     256
#define KOUT    256
#define GH      100
#define GW      100
#define NPIX    (GH*GW)          // 10000
#define ANUM    9
#define NTOT    (NPIX*ANUM)      // 90000
#define IMW     800.0f
#define IMH     800.0f
#define STRD    8                // 800/100
#define SORT_N  131072           // pow2 >= 90000
#define SEG     4096             // LDS-resident sort segment
#define PRENMS  2000
#define POSTK   1000
#define NMS_THR 0.7f
#define BBOX_CLIP 4.135166556742356f   // log(1000/16)
#define KDIM    (CIN*9)          // 2304
#define KCHUNKS (KDIM/32)        // 72
#define XTILES  7                // ceil(100/16)
#define BPITCH  20               // LDS row pitch (floats): 16B-aligned rows, 2-way banks

// LDS byte offset (relative to workgroup LDS base) of a shared-memory pointer
__device__ __forceinline__ unsigned lds_offset_of(const void* p) {
  return (unsigned)(unsigned long long)(const __attribute__((address_space(3))) char*)p;
}

// ---------------------------------------------------------------------------
// Kernel 1: pack conv weights [256,256,3,3] f32 -> bf16 in WMMA A-matrix layout
// Apack[mo][kc][lane][e], k-dim order = tap*256 + c  (tap = dy*3+dx)
// A 16x32 bf16 layout: lane<16: e<8 -> K=e, e>=8 -> K=16+(e-8); lane>=16: +8
// ---------------------------------------------------------------------------
__global__ __launch_bounds__(256) void pack_weights(const float* __restrict__ w,
                                                    bf16* __restrict__ Apack) {
  int gid = blockIdx.x * 256 + threadIdx.x;          // 16*72*32*16 = 589824
  int e    = gid & 15;
  int r    = gid >> 4;
  int lane = r & 31;
  int r2   = r >> 5;
  int kc   = r2 % KCHUNKS;
  int mo   = r2 / KCHUNKS;
  if (mo >= 16) return;
  int hi   = lane >> 4;
  int M    = lane & 15;
  int K32  = (e & 7) + ((e >> 3) << 4) + (hi << 3);  // K within 32-chunk
  int tap  = kc >> 3;
  int c    = (kc & 7) * 32 + K32;
  int kout = mo * 16 + M;
  int dy   = tap / 3, dx = tap % 3;
  float v = w[(((size_t)kout * CIN + c) * 3 + dy) * 3 + dx];
  Apack[gid] = (bf16)v;
}

// ---------------------------------------------------------------------------
// Kernel 2: 3x3 conv (SAME) + bias + ReLU, implicit-GEMM bf16 WMMA.
// Double-buffered: async global->LDS staging of tap t+1 overlaps WMMAs of tap t.
// One block = (b, y, xtile) x 256 out channels; 8 waves x 2 M-tiles each.
// ---------------------------------------------------------------------------
__global__ __launch_bounds__(256) void conv3x3_wmma(const float* __restrict__ feat,
                                                    const bf16*  __restrict__ Apack,
                                                    const float* __restrict__ conv_b,
                                                    bf16* __restrict__ t) {
  __shared__ __align__(16) float Bs[2][CIN * BPITCH];   // 2 x 20KB f32 tap buffers

  const int blk  = blockIdx.x;                  // 0..5599
  const int xt   = blk % XTILES;
  const int y    = (blk / XTILES) % GH;
  const int b    = blk / (XTILES * GH);
  const int x0   = xt * 16;
  const int tid  = threadIdx.x;
  const int lane = tid & 31;
  const int wave = tid >> 5;

  const float* featb = feat + (size_t)b * CIN * NPIX;
  const unsigned lbase[2] = { lds_offset_of(&Bs[0][0]), lds_offset_of(&Bs[1][0]) };

  // stage one tap (256 ch x 16 px) into buffer `buf`; one channel per thread.
  auto stage_tap = [&](int buf, int tap) {
    const int dy = tap / 3 - 1;
    const int dx = tap % 3 - 1;
    const int ay = y + dy;
    const bool yok = (ay >= 0) && (ay < GH);
    const int ch = tid;
    const int rowoff = ch * NPIX + ay * GW;     // element offset of row start
    const unsigned lrow = lbase[buf] + (unsigned)(ch * BPITCH * 4);
    float* bp = &Bs[buf][ch * BPITCH];
    #pragma unroll
    for (int e = 0; e < 16; ++e) {
      const int ax = x0 + dx + e;
      if (yok && ax >= 0 && ax < GW) {
        unsigned lds  = lrow + (unsigned)(e * 4);
        unsigned goff = (unsigned)((rowoff + ax) * 4);
        asm volatile("global_load_async_to_lds_b32 %0, %1, %2"
                     :: "v"(lds), "v"(goff), "s"(featb) : "memory");
      } else {
        bp[e] = 0.0f;                            // boundary zero via normal DS store
      }
    }
  };

  v8f acc0 = {}; v8f acc1 = {};

  stage_tap(0, 0);
  asm volatile("s_wait_asynccnt 0x0" ::: "memory");
  __syncthreads();

  int buf = 0;
  for (int tap = 0; tap < 9; ++tap) {
    if (tap < 8) stage_tap(buf ^ 1, tap + 1);   // async-fill next buffer

    const float* bbase = &Bs[buf][0];
    #pragma unroll
    for (int sub = 0; sub < 8; ++sub) {
      const int kc = tap * 8 + sub;
      // B fragment: lane = pixel (N = lane&15), elements = K rows (channels), table order
      const float* src = bbase + (sub * 32 + ((lane >> 4) << 3)) * BPITCH + (lane & 15);
      v16bf bfrag;
      #pragma unroll
      for (int i = 0; i < 16; ++i) bfrag[i] = (bf16)src[((i & 7) + ((i >> 3) << 4)) * BPITCH];   // f32->bf16 cvt
      const bf16* ap = Apack + ((((size_t)(wave * 2) * KCHUNKS) + kc) * 32 + lane) * 16;
      v16bf a0 = *(const v16bf*)ap;
      v16bf a1 = *(const v16bf*)(ap + (size_t)KCHUNKS * 32 * 16);
      acc0 = __builtin_amdgcn_wmma_f32_16x16x32_bf16(false, a0, false, bfrag,
                                                     (short)0, acc0, false, false);
      acc1 = __builtin_amdgcn_wmma_f32_16x16x32_bf16(false, a1, false, bfrag,
                                                     (short)0, acc1, false, false);
    }

    asm volatile("s_wait_asynccnt 0x0" ::: "memory");  // next buffer fully landed
    __syncthreads();
    buf ^= 1;
  }

  // epilogue: C/D layout -> lane: N = lane&15 (pixel), row r: M = r + 8*(lane>>4)
  const int n  = lane & 15;
  const int hi = lane >> 4;
  const int px = x0 + n;
  if (px < GW) {
    const int pix = y * GW + px;
    #pragma unroll
    for (int r = 0; r < 8; ++r) {
      int k0 = (wave * 2 + 0) * 16 + hi * 8 + r;
      int k1 = (wave * 2 + 1) * 16 + hi * 8 + r;
      float v0 = fmaxf(acc0[r] + conv_b[k0], 0.0f);
      float v1 = fmaxf(acc1[r] + conv_b[k1], 0.0f);
      t[((size_t)b * KOUT + k0) * NPIX + pix] = (bf16)v0;
      t[((size_t)b * KOUT + k1) * NPIX + pix] = (bf16)v1;
    }
  }
}

// ---------------------------------------------------------------------------
// Kernel 3: fused 1x1 heads + sigmoid + anchors + decode + clip + sort keys
// thread = (b, pixel). 45 output channels (9 cls + 36 reg), weights in LDS.
// ---------------------------------------------------------------------------
__global__ __launch_bounds__(256) void rpn_head(const bf16* __restrict__ t,
                                                const float* __restrict__ cls_w,
                                                const float* __restrict__ cls_b,
                                                const float* __restrict__ reg_w,
                                                const float* __restrict__ reg_b,
                                                float* __restrict__ boxes,
                                                unsigned long long* __restrict__ keys) {
  __shared__ float wS[45 * CIN];
  __shared__ float bS[45];
  const int tid = threadIdx.x;
  for (int i = tid; i < 45 * CIN; i += 256)
    wS[i] = (i < 9 * CIN) ? cls_w[i] : reg_w[i - 9 * CIN];
  if (tid < 45) bS[tid] = (tid < 9) ? cls_b[tid] : reg_b[tid - 9];
  __syncthreads();

  const int gid = blockIdx.x * 256 + tid;
  if (gid >= BATCH * NPIX) return;
  const int b   = gid / NPIX;
  const int pix = gid % NPIX;
  const int py_ = pix / GW;
  const int px_ = pix % GW;

  float acc[45];
  #pragma unroll
  for (int o = 0; o < 45; ++o) acc[o] = bS[o];

  const bf16* tb = t + (size_t)b * CIN * NPIX + pix;
  for (int c = 0; c < CIN; ++c) {
    float tv = (float)tb[(size_t)c * NPIX];
    #pragma unroll
    for (int o = 0; o < 45; ++o) acc[o] += tv * wS[o * CIN + c];  // uniform LDS bcast
  }

  const float ratios[3] = {0.5f, 1.0f, 2.0f};
  const float scales[3] = {64.0f, 128.0f, 256.0f};
  #pragma unroll
  for (int a = 0; a < ANUM; ++a) {
    float logit = acc[a];
    float score = 1.0f / (1.0f + __expf(-logit));
    int ri = a / 3, si = a % 3;
    float hr = sqrtf(ratios[ri]);
    float wr = 1.0f / hr;
    float ws_ = wr * scales[si];
    float hs_ = hr * scales[si];
    float ax1 = (float)(px_ * STRD) + rintf(-ws_ * 0.5f);   // banker's rounding
    float ay1 = (float)(py_ * STRD) + rintf(-hs_ * 0.5f);
    float ax2 = (float)(px_ * STRD) + rintf( ws_ * 0.5f);
    float ay2 = (float)(py_ * STRD) + rintf( hs_ * 0.5f);
    float aw = ax2 - ax1, ah = ay2 - ay1;
    float cx = ax1 + 0.5f * aw, cy = ay1 + 0.5f * ah;
    float d0 = acc[9 + a * 4 + 0];
    float d1 = acc[9 + a * 4 + 1];
    float d2 = fminf(acc[9 + a * 4 + 2], BBOX_CLIP);
    float d3 = fminf(acc[9 + a * 4 + 3], BBOX_CLIP);
    float pcx = d0 * aw + cx, pcy = d1 * ah + cy;
    float pw  = __expf(d2) * aw, ph = __expf(d3) * ah;
    float x1 = fminf(fmaxf(pcx - 0.5f * pw, 0.0f), IMW);
    float y1 = fminf(fmaxf(pcy - 0.5f * ph, 0.0f), IMH);
    float x2 = fminf(fmaxf(pcx + 0.5f * pw, 0.0f), IMW);
    float y2 = fminf(fmaxf(pcy + 0.5f * ph, 0.0f), IMH);

    const int nidx = pix * ANUM + a;
    float* bp = boxes + ((size_t)b * NTOT + nidx) * 4;
    bp[0] = x1; bp[1] = y1; bp[2] = x2; bp[3] = y2;
    unsigned int sb = __float_as_uint(score);       // score in (0,1): positive
    keys[(size_t)b * SORT_N + nidx] =
        ((unsigned long long)(~sb) << 32) | (unsigned int)nidx;
  }
}

// pad keys [NTOT, SORT_N) with +inf keys
__global__ __launch_bounds__(256) void pad_keys(unsigned long long* __restrict__ keys) {
  int gid = blockIdx.x * 256 + threadIdx.x;
  int per = SORT_N - NTOT;                         // 41072
  if (gid >= BATCH * per) return;
  int b = gid / per, i = gid % per;
  keys[(size_t)b * SORT_N + NTOT + i] = ~0ull;
}

// ---------------------------------------------------------------------------
// Bitonic sort: LDS-resident kernels fuse all j<=2048 passes (21 launches total)
// ---------------------------------------------------------------------------
__device__ __forceinline__ void cmpswap(unsigned long long* S, int i, int l, bool up) {
  unsigned long long a = S[i], c = S[l];
  if ((a > c) == up) { S[i] = c; S[l] = a; }
}

// full bitonic network k=2..4096 inside one 4096-key segment
__global__ __launch_bounds__(1024) void bitonic_local_sort(unsigned long long* __restrict__ keys) {
  __shared__ unsigned long long S[SEG];
  const int seg = blockIdx.x;                 // 256 segments = 8 images * 32
  const size_t base = (size_t)seg * SEG;
  const int tid = threadIdx.x;
  #pragma unroll
  for (int it = 0; it < 4; ++it) S[tid + it * 1024] = keys[base + tid + it * 1024];
  __syncthreads();
  for (int k = 2; k <= SEG; k <<= 1) {
    for (int j = k >> 1; j >= 1; j >>= 1) {
      #pragma unroll
      for (int it = 0; it < 2; ++it) {
        int p = tid + it * 1024;              // pair id 0..2047
        int i = (p / j) * (2 * j) + (p % j);
        bool up = (((base + (size_t)i) & (size_t)k) == 0);
        cmpswap(S, i, i + j, up);
      }
      __syncthreads();
    }
  }
  #pragma unroll
  for (int it = 0; it < 4; ++it) keys[base + tid + it * 1024] = S[tid + it * 1024];
}

// merge steps j=2048..1 for a given k (k >= 8192)
__global__ __launch_bounds__(1024) void bitonic_local_merge(unsigned long long* __restrict__ keys,
                                                            int k) {
  __shared__ unsigned long long S[SEG];
  const int seg = blockIdx.x;
  const size_t base = (size_t)seg * SEG;
  const int tid = threadIdx.x;
  #pragma unroll
  for (int it = 0; it < 4; ++it) S[tid + it * 1024] = keys[base + tid + it * 1024];
  __syncthreads();
  const bool up = ((base & (size_t)k) == 0);   // uniform: k > segment span
  for (int j = 2048; j >= 1; j >>= 1) {
    #pragma unroll
    for (int it = 0; it < 2; ++it) {
      int p = tid + it * 1024;
      int i = (p / j) * (2 * j) + (p % j);
      cmpswap(S, i, i + j, up);
    }
    __syncthreads();
  }
  #pragma unroll
  for (int it = 0; it < 4; ++it) keys[base + tid + it * 1024] = S[tid + it * 1024];
}

// one global compare-exchange pass (j >= 4096)
__global__ __launch_bounds__(256) void bitonic_pass(unsigned long long* __restrict__ keys,
                                                    int k, int j) {
  int gid = blockIdx.x * 256 + threadIdx.x;        // 8 * 131072 threads
  int b = gid >> 17;
  int i = gid & (SORT_N - 1);
  int l = i ^ j;
  if (l > i) {
    unsigned long long* kb = keys + (size_t)b * SORT_N;
    unsigned long long a = kb[i], c = kb[l];
    bool up = ((i & k) == 0);
    if ((a > c) == up) { kb[i] = c; kb[l] = a; }
  }
}

// ---------------------------------------------------------------------------
// Kernel 4: greedy NMS on top-2000 + stable compaction to [1000,5] output
// ---------------------------------------------------------------------------
__global__ __launch_bounds__(256) void nms_output(const unsigned long long* __restrict__ keys,
                                                  const float* __restrict__ boxes,
                                                  float* __restrict__ out) {
  __shared__ float bxs[PRENMS * 4];
  __shared__ float areas[PRENMS];
  __shared__ float scs[PRENMS];
  __shared__ unsigned char keep8[PRENMS];
  const int b = blockIdx.x;
  const int tid = threadIdx.x;
  const unsigned long long* kb = keys + (size_t)b * SORT_N;
  const float* bb = boxes + (size_t)b * NTOT * 4;

  for (int i = tid; i < PRENMS; i += 256) {
    unsigned long long key = kb[i];
    unsigned int idx = (unsigned int)key;
    scs[i] = __uint_as_float(~(unsigned int)(key >> 32));
    float x1 = bb[idx * 4 + 0], y1 = bb[idx * 4 + 1];
    float x2 = bb[idx * 4 + 2], y2 = bb[idx * 4 + 3];
    bxs[i * 4 + 0] = x1; bxs[i * 4 + 1] = y1;
    bxs[i * 4 + 2] = x2; bxs[i * 4 + 3] = y2;
    areas[i] = (x2 - x1) * (y2 - y1);
    keep8[i] = 1;
  }
  __syncthreads();

  for (int i = 0; i < PRENMS - 1; ++i) {
    if (keep8[i]) {
      float xi1 = bxs[i * 4 + 0], yi1 = bxs[i * 4 + 1];
      float xi2 = bxs[i * 4 + 2], yi2 = bxs[i * 4 + 3];
      float ai = areas[i];
      for (int jj = i + 1 + tid; jj < PRENMS; jj += 256) {
        if (!keep8[jj]) continue;
        float lx = fmaxf(xi1, bxs[jj * 4 + 0]);
        float ly = fmaxf(yi1, bxs[jj * 4 + 1]);
        float rx = fminf(xi2, bxs[jj * 4 + 2]);
        float ry = fminf(yi2, bxs[jj * 4 + 3]);
        float iw = fmaxf(rx - lx, 0.0f);
        float ih = fmaxf(ry - ly, 0.0f);
        float inter = iw * ih;
        float iou = inter / (ai + areas[jj] - inter + 1e-9f);
        if (iou > NMS_THR) keep8[jj] = 0;
      }
    }
    __syncthreads();
  }

  if (tid == 0) {
    float* ob = out + (size_t)b * POSTK * 5;
    int outn = 0;
    for (int i = 0; i < PRENMS && outn < POSTK; ++i) {
      if (keep8[i]) {
        ob[outn * 5 + 0] = bxs[i * 4 + 0];
        ob[outn * 5 + 1] = bxs[i * 4 + 1];
        ob[outn * 5 + 2] = bxs[i * 4 + 2];
        ob[outn * 5 + 3] = bxs[i * 4 + 3];
        ob[outn * 5 + 4] = scs[i];
        ++outn;
      }
    }
    for (int i = 0; i < PRENMS && outn < POSTK; ++i) {
      if (!keep8[i]) {
        ob[outn * 5 + 0] = bxs[i * 4 + 0];
        ob[outn * 5 + 1] = bxs[i * 4 + 1];
        ob[outn * 5 + 2] = bxs[i * 4 + 2];
        ob[outn * 5 + 3] = bxs[i * 4 + 3];
        ob[outn * 5 + 4] = -1.0f;
        ++outn;
      }
    }
  }
}

// ---------------------------------------------------------------------------
extern "C" void kernel_launch(void* const* d_in, const int* in_sizes, int n_in,
                              void* d_out, int out_size, void* d_ws, size_t ws_size,
                              hipStream_t stream) {
  // inputs: 0 image, 1 feat, 2 conv_w, 3 conv_b, 4 cls_w, 5 cls_b, 6 reg_w, 7 reg_b
  const float* feat   = (const float*)d_in[1];
  const float* conv_w = (const float*)d_in[2];
  const float* conv_b = (const float*)d_in[3];
  const float* cls_w  = (const float*)d_in[4];
  const float* cls_b  = (const float*)d_in[5];
  const float* reg_w  = (const float*)d_in[6];
  const float* reg_b  = (const float*)d_in[7];
  float* out = (float*)d_out;

  char* ws = (char*)d_ws;
  size_t off = 0;
  auto take = [&](size_t bytes) { size_t o = off; off = (off + bytes + 255) & ~(size_t)255; return o; };
  bf16* Apack = (bf16*)(ws + take((size_t)16 * KCHUNKS * 32 * 16 * sizeof(bf16)));   // 1.18 MB
  bf16* t     = (bf16*)(ws + take((size_t)BATCH * KOUT * NPIX * sizeof(bf16)));      // 41 MB
  float* boxes = (float*)(ws + take((size_t)BATCH * NTOT * 4 * sizeof(float)));      // 11.5 MB
  unsigned long long* keys =
      (unsigned long long*)(ws + take((size_t)BATCH * SORT_N * 8));                  // 8.4 MB
  (void)ws_size; (void)in_sizes; (void)n_in; (void)out_size;

  pack_weights<<<(16 * KCHUNKS * 32 * 16 + 255) / 256, 256, 0, stream>>>(conv_w, Apack);
  conv3x3_wmma<<<BATCH * GH * XTILES, 256, 0, stream>>>(feat, Apack, conv_b, t);
  rpn_head<<<(BATCH * NPIX + 255) / 256, 256, 0, stream>>>(t, cls_w, cls_b, reg_w, reg_b,
                                                           boxes, keys);
  pad_keys<<<(BATCH * (SORT_N - NTOT) + 255) / 256, 256, 0, stream>>>(keys);

  // bitonic sort, LDS-fused: 21 launches instead of 153
  bitonic_local_sort<<<(BATCH * SORT_N) / SEG, 1024, 0, stream>>>(keys);
  for (int k = SEG * 2; k <= SORT_N; k <<= 1) {
    for (int j = k >> 1; j >= SEG; j >>= 1)
      bitonic_pass<<<(BATCH * SORT_N) / 256, 256, 0, stream>>>(keys, k, j);
    bitonic_local_merge<<<(BATCH * SORT_N) / SEG, 1024, 0, stream>>>(keys, k);
  }

  nms_output<<<BATCH, 256, 0, stream>>>(keys, boxes, out);
}
